// JointAttention_86629490360821
// MI455X (gfx1250) — matrix operand order repaired and checked by reading, not compiled, not run on hardware
//
#include <hip/hip_runtime.h>
#include <hip/hip_bf16.h>

// ---------------------------------------------------------------------------
// Joint attention (Flux-style dual stream) for MI455X / gfx1250.
// bf16 WMMA (v_wmma_f32_16x16x32_bf16), f32 accumulate, double-buffered async
// global->LDS staging (global_load_async_to_lds_b128 / s_wait_asynccnt).
// ---------------------------------------------------------------------------

typedef __bf16 bf16;
typedef __attribute__((ext_vector_type(2)))  __bf16 v2bf;
typedef __attribute__((ext_vector_type(4)))  __bf16 v4bf;
typedef __attribute__((ext_vector_type(8)))  __bf16 v8bf;
typedef __attribute__((ext_vector_type(16))) __bf16 v16bf;
typedef __attribute__((ext_vector_type(8)))  float  v8f;

#define WMMA_BF16(a, b, c) \
  __builtin_amdgcn_wmma_f32_16x16x32_bf16(false, (a), false, (b), (short)0, (c), false, false)

static __device__ __forceinline__ v16bf cat16(v8bf lo, v8bf hi) {
  return __builtin_shufflevector(lo, hi, 0,1,2,3,4,5,6,7,8,9,10,11,12,13,14,15);
}
static __device__ __forceinline__ v8bf ld8(const bf16* p) { return *(const v8bf*)p; }

// Async DMA: copy 16B from global to LDS, tracked by ASYNCcnt (bypasses VGPRs).
// Low 32 bits of a generic pointer to a __shared__ object are the LDS byte
// address (ISA: LDS aperture maps addr[31:0] -> LDS offset).
static __device__ __forceinline__ void async_cp16(bf16* lds_dst, const bf16* gsrc) {
  unsigned lds = (unsigned)(unsigned long long)lds_dst;
  asm volatile("global_load_async_to_lds_b128 %0, %1, off"
               :: "v"(lds), "v"(gsrc) : "memory");
}
static __device__ __forceinline__ void async_wait0() {
  asm volatile("s_wait_asynccnt 0" ::: "memory");
}

// ---------------------------------------------------------------------------
// f32 -> bf16 cast, 4 elements per thread (all sizes are multiples of 4)
// ---------------------------------------------------------------------------
__global__ __launch_bounds__(256) void cast_f32_bf16(const float* __restrict__ in,
                                                     bf16* __restrict__ out, long n4) {
  long i = (long)blockIdx.x * 256 + threadIdx.x;
  if (i < n4) {
    float4 f = *(const float4*)(in + i * 4);
    v4bf b = {(bf16)f.x, (bf16)f.y, (bf16)f.z, (bf16)f.w};
    *(v4bf*)(out + i * 4) = b;
  }
}

// ---------------------------------------------------------------------------
// Stacked pair of GEMMs sharing N,K:  rows [0,M1) use (A1,W1,C1), rows above
// use (A2,W2,C2).   C[M][N] = A[M][K] * W[N][K]^T  (bf16 in, f32 accumulate)
// Block tile 128x128x32, 256 threads = 8 waves (4Mx2N), wave = 2x4 WMMA tiles.
// Double-buffered LDS; next tile DMA'd while current tile is computed.
// ---------------------------------------------------------------------------
template <bool OUT_BF16>
__global__ __launch_bounds__(256) void gemm_bt2(const bf16* __restrict__ A1,
                                                const bf16* __restrict__ W1,
                                                void* __restrict__ C1,
                                                const bf16* __restrict__ A2,
                                                const bf16* __restrict__ W2,
                                                void* __restrict__ C2,
                                                int M1, int N, int K) {
  constexpr int LDT = 40;  // 32 + 8 pad (80B rows: 16B aligned, conflict-friendly)
  __shared__ __align__(16) bf16 As[2][128][LDT];
  __shared__ __align__(16) bf16 Bs[2][128][LDT];

  const int tid  = threadIdx.x;
  const int lane = tid & 31, wid = tid >> 5;
  const int l16  = lane & 15, hk = lane >> 4;     // half-wave select
  const int wm   = wid & 3,  wn = wid >> 2;       // 4 x 2 wave grid
  const int bn   = blockIdx.x * 128;

  const bf16* A; const bf16* W; void* C; int bm;
  if ((int)blockIdx.y * 128 < M1) { A = A1; W = W1; C = C1; bm = blockIdx.y * 128; }
  else                            { A = A2; W = W2; C = C2; bm = blockIdx.y * 128 - M1; }

  auto stage = [&](int buf, int kt) {
#pragma unroll
    for (int j = 0; j < 2; ++j) {
      int c = tid + j * 256;                  // 0..511
      int row = c >> 2, c8 = (c & 3) * 8;     // 4 chunks per 32-wide row
      async_cp16(&As[buf][row][c8], A + (size_t)(bm + row) * K + kt + c8);
      async_cp16(&Bs[buf][row][c8], W + (size_t)(bn + row) * K + kt + c8);
    }
  };

  const v8f vzero = {0.f,0.f,0.f,0.f,0.f,0.f,0.f,0.f};
  v8f acc[2][4];
#pragma unroll
  for (int i = 0; i < 2; ++i)
#pragma unroll
    for (int j = 0; j < 4; ++j) acc[i][j] = vzero;

  auto compute = [&](int buf) {
    // A fragment: lane holds row (l16), K = hk*16 + i (matches B fragment)
    v16bf af[2];
#pragma unroll
    for (int mf = 0; mf < 2; ++mf) {
      int r = wm * 32 + mf * 16 + l16;
      af[mf] = cat16(ld8(&As[buf][r][hk * 16]), ld8(&As[buf][r][hk * 16 + 8]));
    }
    // B fragment: lane holds col (l16), K = hk*16 + i (same order as A fragment)
    v16bf bfm[4];
#pragma unroll
    for (int nf = 0; nf < 4; ++nf) {
      int cl = wn * 64 + nf * 16 + l16;
      bfm[nf] = cat16(ld8(&Bs[buf][cl][hk * 16]), ld8(&Bs[buf][cl][hk * 16 + 8]));
    }
#pragma unroll
    for (int mf = 0; mf < 2; ++mf)
#pragma unroll
      for (int nf = 0; nf < 4; ++nf)
        acc[mf][nf] = WMMA_BF16(af[mf], bfm[nf], acc[mf][nf]);
  };

  // 2-deep pipeline, unrolled x2 so buffer indices are compile-time.
  stage(0, 0);
  for (int kt = 0; kt < K; kt += 64) {
    async_wait0();              // tile (kt) resident (issued last step)
    __syncthreads();            // everyone done with buf1, chunks visible
    if (kt + 32 < K) stage(1, kt + 32);
    compute(0);

    async_wait0();              // tile (kt+32) resident
    __syncthreads();
    if (kt + 64 < K) stage(0, kt + 64);
    compute(1);
  }

  // Epilogue. C layout: VGPR g, lanes 0-15 -> row g, lanes 16-31 -> row g+8.
#pragma unroll
  for (int mf = 0; mf < 2; ++mf)
#pragma unroll
    for (int nf = 0; nf < 4; ++nf)
#pragma unroll
      for (int g = 0; g < 8; ++g) {
        size_t r = (size_t)(bm + wm * 32 + mf * 16 + g + 8 * hk);
        size_t cidx = (size_t)(bn + wn * 64 + nf * 16 + l16);
        float v = acc[mf][nf][g];
        if (OUT_BF16) ((bf16*)C)[r * N + cidx] = (bf16)v;
        else          ((float*)C)[r * N + cidx] = v;
      }
}

// ---------------------------------------------------------------------------
// Scatter fused QKV output into Q/K/V [H][4608][D] bf16, applying RoPE to
// img q,k. Column o of the QKV GEMM maps to (qkv, h, d) = (o/1536, (o%1536)/128, o%128).
// ---------------------------------------------------------------------------
__global__ __launch_bounds__(256) void scatter_qkv(const bf16* __restrict__ QKV,
                                                   const float* __restrict__ rope,
                                                   bf16* __restrict__ Qall,
                                                   bf16* __restrict__ Kc,
                                                   bf16* __restrict__ Vc,
                                                   int S, int tokOff, int isImg) {
  long p = (long)blockIdx.x * 256 + threadIdx.x;       // one thread per (even,odd) pair
  long total = (long)S * 2304;
  if (p >= total) return;
  int s = (int)(p / 2304);
  int o = (int)(p % 2304) * 2;                          // even column
  float x0 = (float)QKV[(size_t)s * 4608 + o];
  float x1 = (float)QKV[(size_t)s * 4608 + o + 1];
  int qkv = o / 1536, r = o % 1536;
  int h = r / 128, d = r % 128;                         // d even
  if (isImg && qkv < 2) {
    float c  = rope[(size_t)s * 128 + d];
    float sn = rope[(size_t)s * 128 + d + 1];
    float y0 = x0 * c - x1 * sn;
    float y1 = x1 * c + x0 * sn;
    x0 = y0; x1 = y1;
  }
  size_t base = ((size_t)h * 4608 + tokOff + s) * 128 + d;
  bf16* dst = (qkv == 0) ? Qall : (qkv == 1) ? Kc : Vc;
  v2bf pk = {(bf16)x0, (bf16)x1};
  *(v2bf*)&dst[base] = pk;                              // one aligned 4B store
}

// ---------------------------------------------------------------------------
// Flash attention: one block = (head, 128 query rows). 8 waves x 16 rows.
// KV tiles of 32 tokens, double-buffered; D=128.
// Per tile/wave: 8 WMMA (scores) + 8 WMMA (PV).
// ---------------------------------------------------------------------------
__global__ __launch_bounds__(256) void attn_fwd(const bf16* __restrict__ Q,
                                                const bf16* __restrict__ Kc,
                                                const bf16* __restrict__ Vc,
                                                bf16* __restrict__ O,
                                                int S, float scale) {
  constexpr int D = 128, LKS = 136, LVT = 40, LPS = 40;
  __shared__ __align__(16) bf16 Ks[2][32][LKS];    // [token][d]
  __shared__ __align__(16) bf16 Vt[2][D][LVT];     // [d][token]  (transposed)
  __shared__ __align__(16) bf16 Ps[8][16][LPS];    // per-wave P tile [qrow][token]

  const int tid  = threadIdx.x;
  const int lane = tid & 31, wid = tid >> 5;
  const int l16  = lane & 15, hk = lane >> 4;
  const int h     = blockIdx.y;
  const int qbase = blockIdx.x * 128 + wid * 16;

  const bf16* Qh = Q  + (size_t)h * S * D;
  const bf16* Kh = Kc + (size_t)h * S * D;
  const bf16* Vh = Vc + (size_t)h * S * D;

  // Preload the wave's 16xD query tile as 4 A-fragments (K chunks of 32).
  v16bf qf[4];
#pragma unroll
  for (int kb = 0; kb < 4; ++kb) {
    const bf16* qp = Qh + (size_t)(qbase + l16) * D + kb * 32;
    qf[kb] = cat16(ld8(qp + hk * 16), ld8(qp + hk * 16 + 8));
  }

  const v8f vzero = {0.f,0.f,0.f,0.f,0.f,0.f,0.f,0.f};
  v8f acc[8];
  float mrow[8], lrow[8];
#pragma unroll
  for (int f = 0; f < 8; ++f) acc[f] = vzero;
#pragma unroll
  for (int g = 0; g < 8; ++g) { mrow[g] = -3.0e38f; lrow[g] = 0.f; }

  // Stage K (async DMA) and V^T (manual transpose) for one 32-token tile.
  auto stage = [&](int buf, int kv) {
#pragma unroll
    for (int j = 0; j < 2; ++j) {
      int c = tid + j * 256;                // 0..511
      int row = c >> 4, c8 = (c & 15) * 8;  // 16 chunks per 128-wide row
      async_cp16(&Ks[buf][row][c8], Kh + (size_t)(kv + row) * D + c8);
      v8bf vv = ld8(Vh + (size_t)(kv + row) * D + c8);
#pragma unroll
      for (int e = 0; e < 8; ++e) Vt[buf][c8 + e][row] = vv[e];
    }
  };

  auto compute = [&](int buf) {
    // Scores: S16x32 = Q(16xD) x K^T(Dx32) as two 16x16 accumulators.
    v8f s0 = vzero, s1 = vzero;
#pragma unroll
    for (int kb = 0; kb < 4; ++kb) {
      v16bf k0 = cat16(ld8(&Ks[buf][l16][kb * 32 + hk * 16]),
                       ld8(&Ks[buf][l16][kb * 32 + hk * 16 + 8]));
      v16bf k1 = cat16(ld8(&Ks[buf][16 + l16][kb * 32 + hk * 16]),
                       ld8(&Ks[buf][16 + l16][kb * 32 + hk * 16 + 8]));
      s0 = WMMA_BF16(qf[kb], k0, s0);
      s1 = WMMA_BF16(qf[kb], k1, s1);
    }

    // Online softmax. Row of element g is (g + 8*hk); its 32 columns live in
    // the 16 lanes of this half-wave across s0/s1 -> shfl_xor 1,2,4,8 reduce.
    float alpha[8];
#pragma unroll
    for (int g = 0; g < 8; ++g) {
      float v0 = s0[g] * scale, v1 = s1[g] * scale;
      float mx = fmaxf(v0, v1);
      mx = fmaxf(mx, __shfl_xor(mx, 1, 32));
      mx = fmaxf(mx, __shfl_xor(mx, 2, 32));
      mx = fmaxf(mx, __shfl_xor(mx, 4, 32));
      mx = fmaxf(mx, __shfl_xor(mx, 8, 32));
      float mn = fmaxf(mrow[g], mx);
      float a  = __expf(mrow[g] - mn);
      float p0 = __expf(v0 - mn), p1 = __expf(v1 - mn);
      float rs = p0 + p1;
      rs += __shfl_xor(rs, 1, 32);
      rs += __shfl_xor(rs, 2, 32);
      rs += __shfl_xor(rs, 4, 32);
      rs += __shfl_xor(rs, 8, 32);
      lrow[g] = lrow[g] * a + rs;
      mrow[g] = mn;
      alpha[g] = a;
      // C-layout -> LDS (row-major P tile) so we can reload as an A fragment.
      Ps[wid][g + 8 * hk][l16]      = (bf16)p0;
      Ps[wid][g + 8 * hk][16 + l16] = (bf16)p1;
    }
#pragma unroll
    for (int f = 0; f < 8; ++f)
#pragma unroll
      for (int g = 0; g < 8; ++g) acc[f][g] *= alpha[g];

    asm volatile("s_wait_dscnt 0" ::: "memory");  // P stores -> P frag reload

    v16bf pa = cat16(ld8(&Ps[wid][l16][hk * 16]), ld8(&Ps[wid][l16][hk * 16 + 8]));
#pragma unroll
    for (int f = 0; f < 8; ++f) {
      v16bf vf = cat16(ld8(&Vt[buf][f * 16 + l16][hk * 16]),
                       ld8(&Vt[buf][f * 16 + l16][hk * 16 + 8]));
      acc[f] = WMMA_BF16(pa, vf, acc[f]);
    }
  };

  // 2-deep pipeline over KV tiles, unrolled x2 (S/32 is even).
  stage(0, 0);
  for (int kv = 0; kv < S; kv += 64) {
    async_wait0();              // tile (kv) resident
    __syncthreads();            // all waves done with buf1; chunks visible
    if (kv + 32 < S) stage(1, kv + 32);
    compute(0);

    async_wait0();              // tile (kv+32) resident
    __syncthreads();
    if (kv + 64 < S) stage(0, kv + 64);
    compute(1);
  }

  // Normalize and emit O[token][h*128 + d] in bf16 (feeds out-proj GEMM).
#pragma unroll
  for (int g = 0; g < 8; ++g) {
    float inv = 1.0f / lrow[g];
    size_t r = (size_t)(qbase + g + 8 * hk);
#pragma unroll
    for (int f = 0; f < 8; ++f)
      O[r * (size_t)1536 + h * 128 + f * 16 + l16] = (bf16)(acc[f][g] * inv);
  }
}

// ---------------------------------------------------------------------------
// Host side
// ---------------------------------------------------------------------------
extern "C" void kernel_launch(void* const* d_in, const int* in_sizes, int n_in,
                              void* d_out, int out_size, void* d_ws, size_t ws_size,
                              hipStream_t stream) {
  (void)in_sizes; (void)n_in; (void)out_size; (void)ws_size;
  const float* txt  = (const float*)d_in[0];   // [1,512,1536]
  const float* img  = (const float*)d_in[1];   // [1,4096,1536]
  const float* rope = (const float*)d_in[2];   // [1,4096,128]
  const float* Wtq  = (const float*)d_in[3];   // [4608,1536]
  const float* Wiq  = (const float*)d_in[4];   // [4608,1536]
  const float* Wto  = (const float*)d_in[5];   // [1536,1536]
  const float* Wio  = (const float*)d_in[6];   // [1536,1536]
  float* out = (float*)d_out;                  // txt_out [512,1536] ++ img_out [4096,1536]

  char* ws = (char*)d_ws;
  size_t off = 0;
  auto alloc = [&](size_t elems) {
    bf16* p = (bf16*)(ws + off);
    off += ((elems * sizeof(bf16) + 255) / 256) * 256;
    return p;
  };
  bf16* Xt   = alloc((size_t)512 * 1536);
  bf16* Xi   = alloc((size_t)4096 * 1536);
  bf16* Bqt  = alloc((size_t)4608 * 1536);
  bf16* Bqi  = alloc((size_t)4608 * 1536);
  bf16* Bot  = alloc((size_t)1536 * 1536);
  bf16* Boi  = alloc((size_t)1536 * 1536);
  bf16* QKVt = alloc((size_t)512 * 4608);
  bf16* QKVi = alloc((size_t)4096 * 4608);
  bf16* Qall = alloc((size_t)12 * 4608 * 128);
  bf16* Kc   = alloc((size_t)12 * 4608 * 128);
  bf16* Vc   = alloc((size_t)12 * 4608 * 128);
  bf16* Obuf = alloc((size_t)4608 * 1536);

  auto cg = [](long n) { return dim3((unsigned)((n + 255) / 256)); };

  cast_f32_bf16<<<cg((long)512 * 1536 / 4),  256, 0, stream>>>(txt, Xt, (long)512 * 1536 / 4);
  cast_f32_bf16<<<cg((long)4096 * 1536 / 4), 256, 0, stream>>>(img, Xi, (long)4096 * 1536 / 4);
  cast_f32_bf16<<<cg((long)4608 * 1536 / 4), 256, 0, stream>>>(Wtq, Bqt, (long)4608 * 1536 / 4);
  cast_f32_bf16<<<cg((long)4608 * 1536 / 4), 256, 0, stream>>>(Wiq, Bqi, (long)4608 * 1536 / 4);
  cast_f32_bf16<<<cg((long)1536 * 1536 / 4), 256, 0, stream>>>(Wto, Bot, (long)1536 * 1536 / 4);
  cast_f32_bf16<<<cg((long)1536 * 1536 / 4), 256, 0, stream>>>(Wio, Boi, (long)1536 * 1536 / 4);

  // QKV projections: txt rows [0,512) + img rows [512,4608), one launch.
  gemm_bt2<true><<<dim3(36, 36), 256, 0, stream>>>(Xt, Bqt, QKVt,
                                                   Xi, Bqi, QKVi,
                                                   512, 4608, 1536);

  // Split into Q/K/V [H][4608][D], RoPE on img q/k
  scatter_qkv<<<cg((long)512 * 2304),  256, 0, stream>>>(QKVt, nullptr, Qall, Kc, Vc, 512,  0,   0);
  scatter_qkv<<<cg((long)4096 * 2304), 256, 0, stream>>>(QKVi, rope,    Qall, Kc, Vc, 4096, 512, 1);

  // Attention over all 4608 queries x 4608 keys, 12 heads
  attn_fwd<<<dim3(36, 12), 256, 0, stream>>>(Qall, Kc, Vc, Obuf, 4608, 0.08838834764831845f);

  // Output projections straight into d_out (f32), one stacked launch.
  gemm_bt2<false><<<dim3(12, 36), 256, 0, stream>>>(Obuf, Bot, out,
                                                    Obuf + (size_t)512 * 1536, Boi,
                                                    out + (size_t)512 * 1536,
                                                    512, 1536, 1536);
}
